// Fingerprint_53841710023395
// MI455X (gfx1250) — matrix operand order repaired and checked by reading, not compiled, not run on hardware
//
#include <hip/hip_runtime.h>
#include <hip/hip_bf16.h>
#include <cmath>

typedef __attribute__((ext_vector_type(16))) _Float16     v16h;
typedef __attribute__((ext_vector_type(8)))  float        v8f;
typedef __attribute__((ext_vector_type(4)))  unsigned int v4u;
typedef __attribute__((ext_vector_type(8)))  int          v8i;
typedef __attribute__((ext_vector_type(4)))  int          v4i;

#define FD        64
#define ATOM_DIM  39
#define BOND_DIM  10
#define N_ATOM    20000
#define N_BOND    40000
#define N_MOL     1000
#define NEG_INF   (-3.4e38f)

#define N_EGRP    (N_BOND / 16)      // 2500 edge groups of 16
#define WPB       8                  // waves per block in k_edge
#define TAB_WENC_DW 16384            // 4096*16 halves as dwords
#define TAB_CFE_DW  4096             // folded bias floats
#define TAB_DW      (TAB_WENC_DW + TAB_CFE_DW)   // 20480 dwords = 80KB

// ---------------------------------------------------------------- helpers
__device__ __forceinline__ float eluf(float x)     { return x > 0.f ? x : (expf(x) - 1.f); }
__device__ __forceinline__ float sigmoidf_(float x){ return 1.f / (1.f + expf(-x)); }
__device__ __forceinline__ float bn_scale(float g) { return g * rsqrtf(1.0f + 1e-6f); }

__device__ __forceinline__ void atomicMaxF(float* a, float v) {
    if (v >= 0.f) atomicMax((int*)a, __float_as_int(v));
    else          atomicMin((unsigned int*)a, __float_as_uint(v));
}

// ---------------------------------------------------------------- fill
__global__ void k_fill(float* p, float v, int n) {
    int i = blockIdx.x * blockDim.x + threadIdx.x;
    if (i < n) p[i] = v;
}

// ---------------------------------------------------------------- weight packing (fold BN into f16 weights)
__global__ void k_pack_enc(const float* __restrict__ w, const float* __restrict__ b,
                           const float* __restrict__ g, const float* __restrict__ bt,
                           _Float16* __restrict__ wp, float* __restrict__ cfe) {
    int j = blockIdx.x * blockDim.x + threadIdx.x;
    if (j >= FD * FD) return;
    float s = bn_scale(g[j]);
#pragma unroll
    for (int k = 0; k < 16; ++k)
        wp[j * 16 + k] = (k < BOND_DIM) ? (_Float16)(w[j * BOND_DIM + k] * s) : (_Float16)0.f;
    cfe[j] = b[j] * s + bt[j];
}

__global__ void k_pack_att(const float* __restrict__ w, const float* __restrict__ b,
                           const float* __restrict__ g, const float* __restrict__ bt,
                           _Float16* __restrict__ wp, float* __restrict__ cfa) {
    int j = threadIdx.x;
    if (j >= FD) return;
    float s = bn_scale(g[j]);
    for (int k = 0; k < FD; ++k) wp[j * FD + k] = (_Float16)(w[j * FD + k] * s);
    cfa[j] = b[j] * s + bt[j];
}

// ---------------------------------------------------------------- preprocess: a = relu(bn(atom @ Wp^T))
__global__ void k_pre(const float* __restrict__ atom,
                      const float* __restrict__ w, const float* __restrict__ b,
                      const float* __restrict__ g, const float* __restrict__ bt,
                      float* __restrict__ out) {
    int t = blockIdx.x * blockDim.x + threadIdx.x;
    if (t >= N_ATOM * FD) return;
    int row = t >> 6, c = t & 63;
    float y = b[c];
    const float* x = atom + (size_t)row * ATOM_DIM;
    const float* wr = w + (size_t)c * ATOM_DIM;
    for (int k = 0; k < ATOM_DIM; ++k) y += x[k] * wr[k];
    y = y * bn_scale(g[c]) + bt[c];
    out[t] = y > 0.f ? y : 0.f;
}

// ---------------------------------------------------------------- fused edge kernel (WMMA + LDS-resident weights)
// Block = 8 waves; the 80KB folded encoder table (f16 weights + f32 bias) is
// DMA'd once per block into LDS by the Tensor Data Mover (fallback: coop copy).
// Each wave handles 16 edges: encoder GEMM via v_wmma_f32_16x16x32_f16 (bias in
// accumulator), fused einsum, attend GEMM via WMMA, score + atomic max.
__global__ __launch_bounds__(256) void k_edge(
    const float* __restrict__ a,
    const float* __restrict__ bond,
    const int*   __restrict__ bidx,
    const _Float16* __restrict__ wencP,  // [4096][16] f16 (CFE f32[4096] contiguous after)
    const float* __restrict__ cfe,       // == (float*)(wencP) + TAB_WENC_DW
    const _Float16* __restrict__ watP,   // [64][64] f16, BN-folded
    const float* __restrict__ cfa,       // [64] folded bias
    const float* __restrict__ al_w,      // [128]
    const float* __restrict__ al_b,      // [1]
    float* __restrict__ att_ws,          // [N_BOND][64]
    float* __restrict__ score_ws,        // [N_BOND]
    float* __restrict__ mmax)            // [N_ATOM] (-inf init)
{
    __shared__ unsigned int tab[TAB_DW];      // 80KB: wenc f16 table then cfe f32
    __shared__ float xs[WPB][16 * 64];        // a[nbr] per wave (broadcast reads)
    __shared__ float nb[WPB][16 * 72];        // neighbor per wave (padded)
    __shared__ int   sidx[WPB][16], nidx[WPB][16];

    const int tid  = threadIdx.x;
    const int lane = tid & 31;
    const int wid  = tid >> 5;
    const int col  = lane & 15;
    const bool hi  = lane >= 16;

    // ---- stage weight table into LDS (once per block)
#if __has_builtin(__builtin_amdgcn_tensor_load_to_lds) && __has_builtin(__builtin_amdgcn_s_wait_tensorcnt)
    if (wid == 0) {
        unsigned long long ga = (unsigned long long)(const void*)wencP;
        unsigned lds_off = (unsigned)(unsigned long long)(uintptr_t)&tab[0];
        v4u g0;
        g0.x = 1u;                                              // count=1 (user descriptor)
        g0.y = lds_off;                                         // lds_addr
        g0.z = (unsigned)(ga & 0xffffffffu);                    // global_addr[31:0]
        g0.w = (unsigned)((ga >> 32) & 0x01ffffffu) | (2u << 30); // global_addr[56:32] | type=2
        v8i g1;
        g1[0] = (2 << 16);                                      // data_size = 4B
        g1[1] = (TAB_DW & 0xffff) << 16;                        // tensor_dim0[15:0]
        g1[2] = ((TAB_DW >> 16) & 0xffff) | (1 << 16);          // tensor_dim0[31:16] | tensor_dim1=1
        g1[3] = (TAB_DW & 0xffff) << 16;                        // tile_dim0 = TAB_DW
        g1[4] = 1;                                              // tile_dim1 = 1
        g1[5] = TAB_DW;                                         // tensor_dim0_stride lo
        g1[6] = 0;
        g1[7] = 0;
        v4i gz4 = {0, 0, 0, 0};
        v8i gz8 = {0, 0, 0, 0, 0, 0, 0, 0};
        __builtin_amdgcn_tensor_load_to_lds(g0, g1, gz4, gz4, gz8, 0);
        __builtin_amdgcn_s_wait_tensorcnt(0);
    }
#else
    {
        const unsigned int* wsrc = (const unsigned int*)wencP;
        for (int i = tid; i < TAB_WENC_DW; i += 256) tab[i] = wsrc[i];
        const unsigned int* csrc = (const unsigned int*)cfe;
        for (int i = tid; i < TAB_CFE_DW; i += 256) tab[TAB_WENC_DW + i] = csrc[i];
    }
#endif

    // edge group for this wave (tail waves clamp -> idempotent duplicate work)
    int g = blockIdx.x * WPB + wid;
    if (g > N_EGRP - 1) g = N_EGRP - 1;
    const int eb = g * 16;

    if (lane < 16) {
        sidx[wid][lane] = bidx[2 * (eb + lane) + 0];
        nidx[wid][lane] = bidx[2 * (eb + lane) + 1];
    }
    __syncthreads();   // table + idx visible

    for (int i = lane; i < 16 * 64; i += 32) {
        int r = i >> 6, c = i & 63;
        xs[wid][r * 64 + c] = a[(size_t)nidx[wid][r] * FD + c];
    }
    __syncthreads();   // xs staged (uniform across block)

    const _Float16* wl = (const _Float16*)tab;
    const float*    cl = (const float*)(tab + TAB_WENC_DW);

    // A operand: bond tile 16x32 f16 (row M = lane&15; lanes<16 carry K=0..15, lanes>=16 K=16..31 == 0)
    v16h Aenc = {};
    {
        const float* bp = bond + (size_t)(eb + col) * BOND_DIM;
        if (!hi) {
#pragma unroll
            for (int k = 0; k < 8; ++k) Aenc[k] = (_Float16)bp[k];
            Aenc[8] = (_Float16)bp[8];
            Aenc[9] = (_Float16)bp[9];
        }
    }

    v8f acc[4];
#pragma unroll
    for (int fb = 0; fb < 4; ++fb) { v8f z = {}; acc[fb] = z; }

    const int rowoff = hi ? 8 * 64 : 0;
    for (int d = 0; d < FD; ++d) {
        float xv[8];
#pragma unroll
        for (int r = 0; r < 8; ++r) xv[r] = xs[wid][rowoff + r * 64 + d];  // LDS broadcast
#pragma unroll
        for (int fb = 0; fb < 4; ++fb) {
            int j = (d * 4 + fb) * 16 + col;  // flat encoder output index, d constant per tile
            v16h B = {};
            if (!hi) B = *(const v16h*)(wl + (size_t)j * 16);  // lanes>=16: K=16..31 == 0
            float cf = cl[j];                                  // lane-constant folded bias
            v8f c0 = {cf, cf, cf, cf, cf, cf, cf, cf};         // bias folded into accumulator
            v8f c = __builtin_amdgcn_wmma_f32_16x16x32_f16(false, Aenc, false, B,
                                                           (short)0, c0, false, false);
#pragma unroll
            for (int r = 0; r < 8; ++r) {
                float bm = c[r] > 0.f ? c[r] : 0.f;   // relu -> bmat element
                acc[fb][r] += xv[r] * bm;             // fused einsum 'ed,edf->ef'
            }
        }
    }

    // stage neighbor (C layout -> LDS row-major)
    {
        int rb = hi ? 8 : 0;
#pragma unroll
        for (int fb = 0; fb < 4; ++fb)
#pragma unroll
            for (int r = 0; r < 8; ++r)
                nb[wid][(r + rb) * 72 + fb * 16 + col] = acc[fb][r];
    }
    __syncthreads();

    // attend = neighbor @ watP^T + cfa via WMMA (A rebuilt in A-layout from LDS)
    v16h Aat[2];
    {
        int koff = hi ? 16 : 0;
#pragma unroll
        for (int kh = 0; kh < 2; ++kh) {
            v16h A;
#pragma unroll
            for (int t = 0; t < 8; ++t) {
                A[t]     = (_Float16)nb[wid][col * 72 + kh * 32 + koff + t];
                A[8 + t] = (_Float16)nb[wid][col * 72 + kh * 32 + koff + 8 + t];
            }
            Aat[kh] = A;
        }
    }
#pragma unroll
    for (int jt = 0; jt < 4; ++jt) {
        int j = jt * 16 + col;
        float cf = cfa[j];
        v8f c = {cf, cf, cf, cf, cf, cf, cf, cf};   // bias folded into accumulator
#pragma unroll
        for (int kh = 0; kh < 2; ++kh) {
            v16h B = *(const v16h*)(watP + (size_t)j * FD + kh * 32 + (hi ? 16 : 0));
            c = __builtin_amdgcn_wmma_f32_16x16x32_f16(false, Aat[kh], false, B,
                                                       (short)0, c, false, false);
        }
        int rb = hi ? 8 : 0;
#pragma unroll
        for (int r = 0; r < 8; ++r)
            att_ws[(size_t)(eb + r + rb) * FD + j] = c[r];
    }

    // score = leaky_relu([a_src, neighbor] . w_align + b)
    if (lane < 16) {
        const float* ap = a + (size_t)sidx[wid][lane] * FD;
        float s = al_b[0];
#pragma unroll 8
        for (int f = 0; f < FD; ++f) s += ap[f] * al_w[f];
#pragma unroll 8
        for (int f = 0; f < FD; ++f) s += nb[wid][lane * 72 + f] * al_w[64 + f];
        s = s > 0.f ? s : 0.01f * s;
        score_ws[eb + lane] = s;
        atomicMaxF(&mmax[sidx[wid][lane]], s);
    }
}

// ---------------------------------------------------------------- segment softmax pass 2: e=exp(x-m), sum
__global__ void k_soft(const int* __restrict__ idx, int stride, int n,
                       const float* __restrict__ sc, const float* __restrict__ m,
                       float* __restrict__ ee, float* __restrict__ ssum) {
    int e = blockIdx.x * blockDim.x + threadIdx.x;
    if (e >= n) return;
    int s = idx[(size_t)e * stride];
    float ex = expf(sc[e] - m[s]);
    ee[e] = ex;
    atomicAdd(&ssum[s], ex);
}

// ---------------------------------------------------------------- ctx scatter: ctx[src] += w * attend
__global__ void k_scatter(const int* __restrict__ bidx, const float* __restrict__ ee,
                          const float* __restrict__ ssum, const float* __restrict__ att,
                          float* __restrict__ ctx) {
    int t = blockIdx.x * blockDim.x + threadIdx.x;
    if (t >= N_BOND * FD) return;
    int e = t >> 6, f = t & 63;
    int s = bidx[2 * e];
    float w = ee[e] / (ssum[s] + 1e-8f);
    atomicAdd(&ctx[(size_t)s * FD + f], w * att[t]);
}

// ---------------------------------------------------------------- GRU: out = (1-z)*n + z*h, x = elu(ctx)
__global__ void k_gru(const float* __restrict__ ctx, const float* __restrict__ h,
                      const float* __restrict__ wih, const float* __restrict__ whh,
                      const float* __restrict__ bih, const float* __restrict__ bhh,
                      float* __restrict__ out, int n) {
    int t = blockIdx.x * blockDim.x + threadIdx.x;
    if (t >= n * FD) return;
    int row = t >> 6, c = t & 63;
    const float* x = ctx + (size_t)row * FD;
    const float* hp = h + (size_t)row * FD;
    float gir = bih[c], giz = bih[64 + c], gin = bih[128 + c];
    float ghr = bhh[c], ghz = bhh[64 + c], ghn = bhh[128 + c];
    for (int f = 0; f < FD; ++f) {
        float xv = eluf(x[f]);
        float hv = hp[f];
        gir += xv * wih[(size_t)c * FD + f];
        giz += xv * wih[(size_t)(64 + c) * FD + f];
        gin += xv * wih[(size_t)(128 + c) * FD + f];
        ghr += hv * whh[(size_t)c * FD + f];
        ghz += hv * whh[(size_t)(64 + c) * FD + f];
        ghn += hv * whh[(size_t)(128 + c) * FD + f];
    }
    float r = sigmoidf_(gir + ghr);
    float z = sigmoidf_(giz + ghz);
    float nn = tanhf(gin + r * ghn);
    out[t] = (1.f - z) * nn + z * hp[c];
}

// ---------------------------------------------------------------- superatom init: segment_sum(a, mol)
__global__ void k_supsum(const float* __restrict__ a, const int* __restrict__ mol,
                         float* __restrict__ sup) {
    int t = blockIdx.x * blockDim.x + threadIdx.x;
    if (t >= N_ATOM * FD) return;
    int row = t >> 6, c = t & 63;
    atomicAdd(&sup[(size_t)mol[row] * FD + c], a[t]);
}

// ---------------------------------------------------------------- super attn score
__global__ void k_sscore(const float* __restrict__ sup, const float* __restrict__ a,
                         const int* __restrict__ mol,
                         const float* __restrict__ al_w, const float* __restrict__ al_b,
                         float* __restrict__ sca, float* __restrict__ ms) {
    int i = blockIdx.x * blockDim.x + threadIdx.x;
    if (i >= N_ATOM) return;
    int m = mol[i];
    const float* se = sup + (size_t)m * FD;
    const float* ar = a + (size_t)i * FD;
    float s = al_b[0];
    for (int f = 0; f < FD; ++f) s += se[f] * al_w[f];
    for (int f = 0; f < FD; ++f) s += ar[f] * al_w[64 + f];
    s = s > 0.f ? s : 0.01f * s;
    sca[i] = s;
    atomicMaxF(&ms[m], s);
}

// ---------------------------------------------------------------- super ctx: ctxs[mol] += w * bn(lin(a))
__global__ void k_sctx(const float* __restrict__ a, const int* __restrict__ mol,
                       const float* __restrict__ ee, const float* __restrict__ ss,
                       const float* __restrict__ at_w, const float* __restrict__ at_b,
                       const float* __restrict__ at_g, const float* __restrict__ at_bt,
                       float* __restrict__ ctxs) {
    int t = blockIdx.x * blockDim.x + threadIdx.x;
    if (t >= N_ATOM * FD) return;
    int row = t >> 6, c = t & 63;
    int m = mol[row];
    const float* ar = a + (size_t)row * FD;
    const float* wr = at_w + (size_t)c * FD;
    float y = at_b[c];
    for (int f = 0; f < FD; ++f) y += ar[f] * wr[f];
    y = y * bn_scale(at_g[c]) + at_bt[c];
    float w = ee[row] / (ss[m] + 1e-8f);
    atomicAdd(&ctxs[(size_t)m * FD + c], w * y);
}

// ---------------------------------------------------------------- prediction head
__global__ void k_pred1(const float* __restrict__ sup,
                        const float* __restrict__ w, const float* __restrict__ b,
                        const float* __restrict__ g, const float* __restrict__ bt,
                        float* __restrict__ H) {
    int t = blockIdx.x * blockDim.x + threadIdx.x;
    if (t >= N_MOL * 512) return;
    int row = t >> 9, u = t & 511;
    const float* sr = sup + (size_t)row * FD;
    const float* wr = w + (size_t)u * FD;
    float y = b[u];
    for (int f = 0; f < FD; ++f) y += sr[f] * wr[f];
    y = y * bn_scale(g[u]) + bt[u];
    H[t] = y > 0.f ? y : 0.f;
}

__global__ void k_pred2(const float* __restrict__ H, const float* __restrict__ w2,
                        const float* __restrict__ b2, float* __restrict__ out) {
    int r = blockIdx.x * blockDim.x + threadIdx.x;
    if (r >= N_MOL) return;
    float y = b2[0];
    const float* hr = H + (size_t)r * 512;
    for (int u = 0; u < 512; ++u) y += hr[u] * w2[u];
    out[r] = y;
}

// ---------------------------------------------------------------- launch
#define GRID(n) dim3((unsigned)(((n) + 255) / 256)), dim3(256)

extern "C" void kernel_launch(void* const* d_in, const int* in_sizes, int n_in,
                              void* d_out, int out_size, void* d_ws, size_t ws_size,
                              hipStream_t stream) {
    (void)in_sizes; (void)n_in; (void)out_size; (void)ws_size;
    const float* atom = (const float*)d_in[0];
    const float* bond = (const float*)d_in[1];
    const int* bidx = (const int*)d_in[2];
    const int* mol  = (const int*)d_in[3];
    // d_in[4] = Sentinel (unused)
    auto F = [&](int i) { return (const float*)d_in[i]; };
    const int P0 = 5;                     // preprocess {w,b,g,bt}
    const int PROP0 = P0 + 4;             // 3 layers x 14 leaves
    const int SUP0 = PROP0 + 3 * 14;      // 3 layers x 10 leaves
    const int PRED = SUP0 + 3 * 10;       // pred1 {w,b,g,bt}, pred2 {w,b}

    // workspace carve-up (~29.2 MB of floats)
    float* ws = (float*)d_ws;
    size_t o = 0;
    auto alloc = [&](size_t n) { float* p = ws + o; o += (n + 63) & ~(size_t)63; return p; };
    float* A0   = alloc((size_t)N_ATOM * FD);
    float* A1   = alloc((size_t)N_ATOM * FD);
    float* CTX  = alloc((size_t)N_ATOM * FD);
    float* ATT  = alloc((size_t)N_BOND * FD);
    float* SC   = alloc(N_BOND);
    float* EE   = alloc(N_BOND);
    float* M    = alloc(N_ATOM);
    float* S    = alloc(N_ATOM);
    // wenc f16 table and folded bias MUST be contiguous (single 80KB TDM tile)
    _Float16* WENCH = (_Float16*)alloc(TAB_WENC_DW);  // 4096x16 halves = 16384 dwords
    float* CFE  = alloc(TAB_CFE_DW);                  // directly after (16384 % 64 == 0)
    float* CFA  = alloc(FD);
    float* SUPA = alloc((size_t)N_MOL * FD);
    float* SUPB = alloc((size_t)N_MOL * FD);
    float* CTXS = alloc((size_t)N_MOL * FD);
    float* MS   = alloc(N_MOL);
    float* SS   = alloc(N_MOL);
    float* SCA  = alloc(N_ATOM);
    float* EEA  = alloc(N_ATOM);
    float* H512 = alloc((size_t)N_MOL * 512);
    _Float16* WATH  = (_Float16*)alloc((size_t)FD * FD / 2);  // 64x64 halves

    // ---- preprocess
    k_pre<<<GRID(N_ATOM * FD), 0, stream>>>(atom, F(P0), F(P0 + 1), F(P0 + 2), F(P0 + 3), A0);

    float* aCur = A0;
    float* aNxt = A1;
    for (int l = 0; l < 3; ++l) {
        int b = PROP0 + l * 14;
        // leaves: enc{w,b,g,bt}=0..3, align{w,b}=4..5, attend{w,b,g,bt}=6..9, gru{wih,whh,bih,bhh}=10..13
        k_pack_enc<<<GRID(FD * FD), 0, stream>>>(F(b), F(b + 1), F(b + 2), F(b + 3), WENCH, CFE);
        k_pack_att<<<dim3(1), dim3(64), 0, stream>>>(F(b + 6), F(b + 7), F(b + 8), F(b + 9), WATH, CFA);
        k_fill<<<GRID(N_ATOM * FD), 0, stream>>>(CTX, 0.f, N_ATOM * FD);
        k_fill<<<GRID(N_ATOM), 0, stream>>>(M, NEG_INF, N_ATOM);
        k_fill<<<GRID(N_ATOM), 0, stream>>>(S, 0.f, N_ATOM);
        k_edge<<<dim3((N_EGRP + WPB - 1) / WPB), dim3(256), 0, stream>>>(
            aCur, bond, bidx, WENCH, CFE, WATH, CFA, F(b + 4), F(b + 5), ATT, SC, M);
        k_soft<<<GRID(N_BOND), 0, stream>>>(bidx, 2, N_BOND, SC, M, EE, S);
        k_scatter<<<GRID(N_BOND * FD), 0, stream>>>(bidx, EE, S, ATT, CTX);
        k_gru<<<GRID(N_ATOM * FD), 0, stream>>>(CTX, aCur, F(b + 10), F(b + 11), F(b + 12), F(b + 13),
                                                aNxt, N_ATOM);
        float* tmp = aCur; aCur = aNxt; aNxt = tmp;
    }

    // ---- superatom = segment_sum(a, mol)
    k_fill<<<GRID(N_MOL * FD), 0, stream>>>(SUPA, 0.f, N_MOL * FD);
    k_supsum<<<GRID(N_ATOM * FD), 0, stream>>>(aCur, mol, SUPA);

    float* sCur = SUPA;
    float* sNxt = SUPB;
    for (int t = 0; t < 3; ++t) {
        int b = SUP0 + t * 10;
        // leaves: align{w,b}=0..1, attend{w,b,g,bt}=2..5, gru{wih,whh,bih,bhh}=6..9
        k_fill<<<GRID(N_MOL), 0, stream>>>(MS, NEG_INF, N_MOL);
        k_fill<<<GRID(N_MOL), 0, stream>>>(SS, 0.f, N_MOL);
        k_fill<<<GRID(N_MOL * FD), 0, stream>>>(CTXS, 0.f, N_MOL * FD);
        k_sscore<<<GRID(N_ATOM), 0, stream>>>(sCur, aCur, mol, F(b), F(b + 1), SCA, MS);
        k_soft<<<GRID(N_ATOM), 0, stream>>>(mol, 1, N_ATOM, SCA, MS, EEA, SS);
        k_sctx<<<GRID(N_ATOM * FD), 0, stream>>>(aCur, mol, EEA, SS,
                                                 F(b + 2), F(b + 3), F(b + 4), F(b + 5), CTXS);
        k_gru<<<GRID(N_MOL * FD), 0, stream>>>(CTXS, sCur, F(b + 6), F(b + 7), F(b + 8), F(b + 9),
                                               sNxt, N_MOL);
        float* tmp = sCur; sCur = sNxt; sNxt = tmp;
    }

    // ---- prediction head
    k_pred1<<<GRID(N_MOL * 512), 0, stream>>>(sCur, F(PRED), F(PRED + 1), F(PRED + 2), F(PRED + 3), H512);
    k_pred2<<<GRID(N_MOL), 0, stream>>>(H512, F(PRED + 4), F(PRED + 5), (float*)d_out);
}
